// ES_RNN_MULT_51041391346281
// MI455X (gfx1250) — matrix operand order repaired and checked by reading, not compiled, not run on hardware
//
#include <hip/hip_runtime.h>
#include <hip/hip_bf16.h>
#include <stdint.h>

// ---------------------------------------------------------------------------
// Problem constants (from reference): F,T,S,W,O,H = 2,2048,24,168,48,384
// ---------------------------------------------------------------------------
#define F_DIM   2
#define T_DIM   2048
#define S_DIM   24
#define W_SEQ   168
#define O_DIM   48
#define H_DIM   384
#define G_DIM   1536          // 4*H
#define N_BATCH 1833          // T - W - O + 1
#define N_PAD   1840          // padded to multiple of 16
#define SFULL   2072          // S + 1 + (T-1)

typedef __attribute__((ext_vector_type(8)))  __bf16 bf16x8;
typedef __attribute__((ext_vector_type(16))) __bf16 bf16x16;
typedef __attribute__((ext_vector_type(8)))  float  f32x8;

static __device__ __forceinline__ __bf16 f2bf(float f) {
  uint32_t u = __builtin_bit_cast(uint32_t, f);
  u += 0x7FFFu + ((u >> 16) & 1u);                 // round-to-nearest-even
  uint16_t h = (uint16_t)(u >> 16);
  return __builtin_bit_cast(__bf16, h);
}

static __device__ __forceinline__ float sigm(float x)   { return 1.0f / (1.0f + __expf(-x)); }
static __device__ __forceinline__ float tanh_f(float x) { return 2.0f * sigm(2.0f * x) - 1.0f; }

union AFrag { bf16x16 v; bf16x8 h[2]; };

static __device__ __forceinline__ f32x8 wmma_bf16(bf16x16 a, bf16x16 b, f32x8 c) {
  // D = A(16x32 bf16) * B(32x16 bf16) + C(16x16 f32)
  return __builtin_amdgcn_wmma_f32_16x16x32_bf16(false, a, false, b, (short)0, c, false, false);
}

static __device__ __forceinline__ void fzero(f32x8& v) {
#pragma unroll
  for (int i = 0; i < 8; ++i) v[i] = 0.0f;
}
static __device__ __forceinline__ void bzero16(bf16x16& v) {
  const __bf16 z = f2bf(0.0f);
#pragma unroll
  for (int i = 0; i < 16; ++i) v[i] = z;
}

// Anti-LICM: an opaque zero produced by a volatile asm inside the timestep
// loop. Adding it to the weight index makes every fragment address formally
// loop-variant (so hoisting the 100+ weight loads out of the 168-step loop --
// which ends in scratch spills -- is illegal), while the base pointer itself
// stays untouched: address-space inference keeps the loads as global_load
// (LOADcnt only, saddr+voffset form), costing one s_add per step.
static __device__ __forceinline__ uint32_t opaque_zero() {
  uint32_t z;
  asm volatile("s_mov_b32 %0, 0" : "=s"(z));
  return z;
}

// ---------------------------------------------------------------------------
// Kernel 1: exponential smoothing scan (2 sequential series; trivial cost)
// ---------------------------------------------------------------------------
__global__ void es_kernel(const float* __restrict__ x,
                          const float* __restrict__ level_sm,
                          const float* __restrict__ seas_sm,
                          const float* __restrict__ init_seas,
                          float* __restrict__ levels,
                          float* __restrict__ seas_full) {
  const int f = threadIdx.x;
  __shared__ float buf[F_DIM][S_DIM];
  if (f < F_DIM) {
    const float alpha = sigm(level_sm[f]);
    const float gamma = sigm(seas_sm[f]);
    for (int s = 0; s < S_DIM; ++s) {
      const float v = __expf(init_seas[f * S_DIM + s]);
      buf[f][s] = v;
      seas_full[f * SFULL + s] = v;                 // seas_full[0..23] = seas0
    }
    const float s0 = buf[f][0];
    seas_full[f * SFULL + S_DIM] = s0;              // seas_full[24] = seas0[0]
    float level = x[f * T_DIM] / s0;
    levels[f * T_DIM] = level;
    int head = 1;                                   // buf0 = [s1..s23, s0]
    for (int i = 1; i < T_DIM; ++i) {
      const float xi = x[f * T_DIM + i];
      const float sv = buf[f][head];
      const float nl = alpha * xi / sv + (1.0f - alpha) * level;
      const float ns = gamma * xi / nl + (1.0f - gamma) * sv;
      buf[f][head] = ns;
      head = (head + 1) % S_DIM;
      levels[f * T_DIM + i] = nl;
      seas_full[f * SFULL + S_DIM + i] = ns;
      level = nl;
    }
  }
}

// ---------------------------------------------------------------------------
// Kernel 2: convert/transpose weights to bf16 column-major:
//   dst[n*Kpad + k] = (k<K) ? bf16(src[k*Nc + n]) : 0
// ---------------------------------------------------------------------------
__global__ void convT_kernel(const float* __restrict__ src, __bf16* __restrict__ dst,
                             int K, int Nc, int Kpad, int total) {
  const int idx = blockIdx.x * blockDim.x + threadIdx.x;
  if (idx >= total) return;
  const int n = idx / Kpad, k = idx % Kpad;
  const float v = (k < K) ? src[(size_t)k * Nc + n] : 0.0f;
  dst[idx] = f2bf(v);
}

// ---------------------------------------------------------------------------
// Kernel 3: build LSTM inputs  inputs[n][t][f] (row-major, f fastest)
// ---------------------------------------------------------------------------
__global__ void build_inputs_kernel(const float* __restrict__ x, const float* __restrict__ noise,
                                    const float* __restrict__ levels, const float* __restrict__ seas,
                                    float* __restrict__ inputs) {
  const int idx = blockIdx.x * blockDim.x + threadIdx.x;
  if (idx >= N_BATCH * W_SEQ) return;
  const int n = idx / W_SEQ, t = idx % W_SEQ;
#pragma unroll
  for (int f = 0; f < F_DIM; ++f) {
    const float v = logf(x[f * T_DIM + n + t] /
                         (levels[f * T_DIM + n + W_SEQ] * seas[f * SFULL + n + t])) +
                    noise[((size_t)f * N_BATCH + n) * W_SEQ + t];
    inputs[((size_t)idx) * F_DIM + f] = v;
  }
}

// ---------------------------------------------------------------------------
// Kernel 4: labels -> d_out second half
// ---------------------------------------------------------------------------
__global__ void build_labels_kernel(const float* __restrict__ x, const float* __restrict__ levels,
                                    const float* __restrict__ seas, float* __restrict__ labels) {
  const int idx = blockIdx.x * blockDim.x + threadIdx.x;
  if (idx >= N_BATCH * O_DIM) return;
  const int n = idx / O_DIM, o = idx % O_DIM;
  const int tpos = n + W_SEQ + o;
  labels[idx] = logf(x[tpos] / (levels[n + W_SEQ] * seas[tpos]));
}

// ---------------------------------------------------------------------------
// Kernel 5: LSTM layer 1 (dilation 1). Persistent 16-row batch tile per block,
// 168 sequential steps. 12 waves (384 threads); each wave owns 32 gate-aligned
// columns per gate = 8 n-tiles. Per k-slice: batch-load 8 B fragments, then
// 8 WMMAs -- source order guarantees 8 loads in flight before the first
// consumer so the waits stagger instead of serializing. Weight loads stay in
// the loop via opaque_zero() (no LICM->spill), as global_load (LOADcnt only).
// Emits h at t%24==23 to h1_sel (bf16) for the dilated layer.
// ---------------------------------------------------------------------------
__global__ void __launch_bounds__(384) lstm1_kernel(
    const float* __restrict__ inputs, const __bf16* __restrict__ WhT,
    const __bf16* __restrict__ WxTp, const float* __restrict__ b0,
    __bf16* __restrict__ h1_sel) {
  const int tid  = threadIdx.x;
  const int wave = tid >> 5;           // 0..11
  const int lane = tid & 31;
  const int l16  = lane & 15;
  const int lh   = lane >> 4;
  const int batch0 = blockIdx.x * 16;
  const int colw = wave * 32;          // this wave's h-column block (per gate)

  __shared__ __attribute__((aligned(32))) __bf16 h_lds[16 * H_DIM];
  for (int i = tid; i < 16 * H_DIM; i += 384) h_lds[i] = f2bf(0.0f);
  __syncthreads();

  float bias[8];
#pragma unroll
  for (int j = 0; j < 8; ++j) bias[j] = b0[(j >> 1) * H_DIM + colw + (j & 1) * 16 + l16];

  f32x8 cfr[2];
#pragma unroll
  for (int s = 0; s < 2; ++s) fzero(cfr[s]);

  const int  ng_a = batch0 + l16;
  const bool a_ok = (ng_a < N_BATCH);

  for (int t = 0; t < W_SEQ; ++t) {
    const uint32_t zo = opaque_zero();   // loop-variant 0: defeats load LICM

    f32x8 acc[8];
#pragma unroll
    for (int j = 0; j < 8; ++j) fzero(acc[j]);

    // ---- input contribution: x_t (16x2) via zero-padded K=32 WMMA ----
    bf16x16 ax;
    bzero16(ax);
    if (lane < 16) {
      float x0 = 0.0f, x1 = 0.0f;
      if (a_ok) {
        const float* xp = inputs + ((size_t)ng_a * W_SEQ + t) * F_DIM;
        x0 = xp[0]; x1 = xp[1];
      }
      ax[0] = f2bf(x0);                // lanes 0-15, elems 0,1 => k=0,1
      ax[1] = f2bf(x1);
    }
    {
      bf16x16 bx[8];
#pragma unroll
      for (int j = 0; j < 8; ++j) {
        const int col0 = (j >> 1) * H_DIM + colw + (j & 1) * 16;
        bx[j] = *(const bf16x16*)(WxTp + (size_t)(col0 + l16) * 32 + lh * 16 + zo);
      }
#pragma unroll
      for (int j = 0; j < 8; ++j) acc[j] = wmma_bf16(ax, bx[j], acc[j]);
    }

    // ---- recurrent contribution: h (16x384) @ Wh (384x1536) ----
    for (int k0 = 0; k0 < H_DIM; k0 += 32) {
      AFrag a;
      const __bf16* hp = h_lds + l16 * H_DIM + k0 + lh * 16;
      a.h[0] = *(const bf16x8*)hp;
      a.h[1] = *(const bf16x8*)(hp + 8);
      bf16x16 bh[8];
#pragma unroll
      for (int j = 0; j < 8; ++j) {
        const int col0 = (j >> 1) * H_DIM + colw + (j & 1) * 16;
        bh[j] = *(const bf16x16*)(WhT + (size_t)(col0 + l16) * H_DIM + k0 + lh * 16 + zo);
      }
#pragma unroll
      for (int j = 0; j < 8; ++j) acc[j] = wmma_bf16(a.v, bh[j], acc[j]);
    }
    __syncthreads();                    // all reads of h_lds done

    // ---- gates (wave-local: i/f/g/o columns are gate-aligned) ----
    const bool sel = ((t % 24) == 23);
    const int  ti  = t / 24;
#pragma unroll
    for (int s = 0; s < 2; ++s) {
#pragma unroll
      for (int v = 0; v < 8; ++v) {
        const float iz = sigm(acc[0 + s][v] + bias[0 + s]);
        const float fz = sigm(acc[2 + s][v] + bias[2 + s]);
        const float gz = tanh_f(acc[4 + s][v] + bias[4 + s]);
        const float oz = sigm(acc[6 + s][v] + bias[6 + s]);
        const float c  = fz * cfr[s][v] + iz * gz;
        cfr[s][v] = c;
        const float h  = oz * tanh_f(c);
        const __bf16 hb = f2bf(h);
        const int mrow = v + 8 * lh;              // C-fragment row
        const int col  = colw + s * 16 + l16;
        h_lds[mrow * H_DIM + col] = hb;
        if (sel) {
          const int ng = batch0 + mrow;
          if (ng < N_BATCH)
            h1_sel[((size_t)ng * 7 + ti) * H_DIM + col] = hb;
        }
      }
    }
    __syncthreads();                    // h_lds updated for next step
  }
}

// ---------------------------------------------------------------------------
// Kernel 6: LSTM layer 2 (dilation 24; only the j=23 subsequence feeds
// h2[:, -1, :]): 7 steps over h1_sel. Per step: K=384 over Wx1 + K=384 Wh1.
// Same 12-wave / 8-tile shape and load/compute phasing as lstm1.
// ---------------------------------------------------------------------------
__global__ void __launch_bounds__(384) lstm2_kernel(
    const __bf16* __restrict__ h1_sel, const __bf16* __restrict__ WxT,
    const __bf16* __restrict__ WhT, const float* __restrict__ b1,
    __bf16* __restrict__ h2_last) {
  const int tid  = threadIdx.x;
  const int wave = tid >> 5;           // 0..11
  const int lane = tid & 31;
  const int l16  = lane & 15;
  const int lh   = lane >> 4;
  const int batch0 = blockIdx.x * 16;
  const int colw = wave * 32;

  __shared__ __attribute__((aligned(32))) __bf16 h_lds[16 * H_DIM];
  for (int i = tid; i < 16 * H_DIM; i += 384) h_lds[i] = f2bf(0.0f);
  __syncthreads();

  float bias[8];
#pragma unroll
  for (int j = 0; j < 8; ++j) bias[j] = b1[(j >> 1) * H_DIM + colw + (j & 1) * 16 + l16];

  f32x8 cfr[2];
#pragma unroll
  for (int s = 0; s < 2; ++s) fzero(cfr[s]);

  const int  ng_a = batch0 + l16;
  const bool a_ok = (ng_a < N_BATCH);

  for (int t = 0; t < 7; ++t) {
    const uint32_t zo = opaque_zero();

    f32x8 acc[8];
#pragma unroll
    for (int j = 0; j < 8; ++j) fzero(acc[j]);

    // ---- x_t = h1_sel[:, t, :]  (K=384 over Wx1^T) ----
    for (int k0 = 0; k0 < H_DIM; k0 += 32) {
      AFrag a;
      bzero16(a.v);
      if (a_ok) {
        const __bf16* p = h1_sel + ((size_t)ng_a * 7 + t) * H_DIM + k0 + lh * 16;
        a.h[0] = *(const bf16x8*)p;
        a.h[1] = *(const bf16x8*)(p + 8);
      }
      bf16x16 bx[8];
#pragma unroll
      for (int j = 0; j < 8; ++j) {
        const int col0 = (j >> 1) * H_DIM + colw + (j & 1) * 16;
        bx[j] = *(const bf16x16*)(WxT + (size_t)(col0 + l16) * H_DIM + k0 + lh * 16 + zo);
      }
#pragma unroll
      for (int j = 0; j < 8; ++j) acc[j] = wmma_bf16(a.v, bx[j], acc[j]);
    }
    // ---- recurrent h @ Wh1 ----
    for (int k0 = 0; k0 < H_DIM; k0 += 32) {
      AFrag a;
      const __bf16* hp = h_lds + l16 * H_DIM + k0 + lh * 16;
      a.h[0] = *(const bf16x8*)hp;
      a.h[1] = *(const bf16x8*)(hp + 8);
      bf16x16 bh[8];
#pragma unroll
      for (int j = 0; j < 8; ++j) {
        const int col0 = (j >> 1) * H_DIM + colw + (j & 1) * 16;
        bh[j] = *(const bf16x16*)(WhT + (size_t)(col0 + l16) * H_DIM + k0 + lh * 16 + zo);
      }
#pragma unroll
      for (int j = 0; j < 8; ++j) acc[j] = wmma_bf16(a.v, bh[j], acc[j]);
    }
    __syncthreads();
#pragma unroll
    for (int s = 0; s < 2; ++s) {
#pragma unroll
      for (int v = 0; v < 8; ++v) {
        const float iz = sigm(acc[0 + s][v] + bias[0 + s]);
        const float fz = sigm(acc[2 + s][v] + bias[2 + s]);
        const float gz = tanh_f(acc[4 + s][v] + bias[4 + s]);
        const float oz = sigm(acc[6 + s][v] + bias[6 + s]);
        const float c  = fz * cfr[s][v] + iz * gz;
        cfr[s][v] = c;
        const float h  = oz * tanh_f(c);
        const __bf16 hb = f2bf(h);
        const int mrow = v + 8 * lh;
        const int col  = colw + s * 16 + l16;
        h_lds[mrow * H_DIM + col] = hb;
        if (t == 6) h2_last[(size_t)(batch0 + mrow) * H_DIM + col] = hb;
      }
    }
    __syncthreads();
  }
}

// ---------------------------------------------------------------------------
// Kernel 7: tmp = tanh(h2 @ Wt + bt)  -> bf16 (N_PAD x 384)
// ---------------------------------------------------------------------------
__global__ void __launch_bounds__(256) proj_tanh_kernel(
    const __bf16* __restrict__ h2, const __bf16* __restrict__ WtT,
    const float* __restrict__ bt, __bf16* __restrict__ tmp) {
  const int tid  = threadIdx.x;
  const int wave = tid >> 5;
  const int lane = tid & 31;
  const int l16  = lane & 15;
  const int lh   = lane >> 4;
  const int batch0 = blockIdx.x * 16;
  const int colw = wave * 48;

  f32x8 acc[3];
#pragma unroll
  for (int s = 0; s < 3; ++s) fzero(acc[s]);

  for (int k0 = 0; k0 < H_DIM; k0 += 32) {
    AFrag a;
    const __bf16* p = h2 + (size_t)(batch0 + l16) * H_DIM + k0 + lh * 16;
    a.h[0] = *(const bf16x8*)p;
    a.h[1] = *(const bf16x8*)(p + 8);
    bf16x16 b[3];
#pragma unroll
    for (int s = 0; s < 3; ++s)
      b[s] = *(const bf16x16*)(WtT + (size_t)(colw + s * 16 + l16) * H_DIM + k0 + lh * 16);
#pragma unroll
    for (int s = 0; s < 3; ++s) acc[s] = wmma_bf16(a.v, b[s], acc[s]);
  }
#pragma unroll
  for (int s = 0; s < 3; ++s) {
    const int col = colw + s * 16 + l16;
    const float bv = bt[col];
#pragma unroll
    for (int v = 0; v < 8; ++v)
      tmp[(size_t)(batch0 + v + 8 * lh) * H_DIM + col] = f2bf(tanh_f(acc[s][v] + bv));
  }
}

// ---------------------------------------------------------------------------
// Kernel 8: out = tmp @ Wl + bl -> f32 (N x 48) into d_out first half
// ---------------------------------------------------------------------------
__global__ void __launch_bounds__(32) proj_out_kernel(
    const __bf16* __restrict__ tmp, const __bf16* __restrict__ WlT,
    const float* __restrict__ bl, float* __restrict__ out) {
  const int lane = threadIdx.x & 31;
  const int l16  = lane & 15;
  const int lh   = lane >> 4;
  const int batch0 = blockIdx.x * 16;

  f32x8 acc[3];
#pragma unroll
  for (int s = 0; s < 3; ++s) fzero(acc[s]);

  for (int k0 = 0; k0 < H_DIM; k0 += 32) {
    AFrag a;
    const __bf16* p = tmp + (size_t)(batch0 + l16) * H_DIM + k0 + lh * 16;
    a.h[0] = *(const bf16x8*)p;
    a.h[1] = *(const bf16x8*)(p + 8);
    bf16x16 b[3];
#pragma unroll
    for (int s = 0; s < 3; ++s)
      b[s] = *(const bf16x16*)(WlT + (size_t)(s * 16 + l16) * H_DIM + k0 + lh * 16);
#pragma unroll
    for (int s = 0; s < 3; ++s) acc[s] = wmma_bf16(a.v, b[s], acc[s]);
  }
#pragma unroll
  for (int s = 0; s < 3; ++s) {
    const float bv = bl[s * 16 + l16];
#pragma unroll
    for (int v = 0; v < 8; ++v) {
      const int ng = batch0 + v + 8 * lh;
      if (ng < N_BATCH) out[(size_t)ng * O_DIM + s * 16 + l16] = acc[s][v] + bv;
    }
  }
}

// ---------------------------------------------------------------------------
// Host launcher
// ---------------------------------------------------------------------------
extern "C" void kernel_launch(void* const* d_in, const int* in_sizes, int n_in,
                              void* d_out, int out_size, void* d_ws, size_t ws_size,
                              hipStream_t stream) {
  const float* x         = (const float*)d_in[0];
  const float* noise     = (const float*)d_in[1];
  const float* level_sm  = (const float*)d_in[2];
  const float* seas_sm   = (const float*)d_in[3];
  const float* init_seas = (const float*)d_in[4];
  const float* Wx0       = (const float*)d_in[5];
  const float* Wh0       = (const float*)d_in[6];
  const float* b0        = (const float*)d_in[7];
  const float* Wx1       = (const float*)d_in[8];
  const float* Wh1       = (const float*)d_in[9];
  const float* b1        = (const float*)d_in[10];
  const float* Wt        = (const float*)d_in[11];
  const float* bt        = (const float*)d_in[12];
  const float* Wl        = (const float*)d_in[13];
  const float* bl        = (const float*)d_in[14];
  (void)in_sizes; (void)n_in; (void)out_size; (void)ws_size;

  char* ws = (char*)d_ws;
  size_t off = 0;
  auto take = [&](size_t bytes) -> char* {
    char* p = ws + off;
    off = (off + bytes + 255) & ~(size_t)255;
    return p;
  };

  float*  levels  = (float*)take((size_t)F_DIM * T_DIM * 4);
  float*  seas    = (float*)take((size_t)F_DIM * SFULL * 4);
  float*  inputs  = (float*)take((size_t)N_BATCH * W_SEQ * F_DIM * 4);
  __bf16* Wh0T    = (__bf16*)take((size_t)G_DIM * H_DIM * 2);
  __bf16* Wx0Tp   = (__bf16*)take((size_t)G_DIM * 32 * 2);
  __bf16* Wx1T    = (__bf16*)take((size_t)G_DIM * H_DIM * 2);
  __bf16* Wh1T    = (__bf16*)take((size_t)G_DIM * H_DIM * 2);
  __bf16* WtT     = (__bf16*)take((size_t)H_DIM * H_DIM * 2);
  __bf16* WlT     = (__bf16*)take((size_t)O_DIM * H_DIM * 2);
  __bf16* h1_sel  = (__bf16*)take((size_t)N_PAD * 7 * H_DIM * 2);
  __bf16* h2_last = (__bf16*)take((size_t)N_PAD * H_DIM * 2);
  __bf16* tmp     = (__bf16*)take((size_t)N_PAD * H_DIM * 2);

  float* out    = (float*)d_out;
  float* labels = out + (size_t)N_BATCH * O_DIM;

  es_kernel<<<1, 32, 0, stream>>>(x, level_sm, seas_sm, init_seas, levels, seas);

  auto conv = [&](const float* src, __bf16* dst, int K, int Nc, int Kpad) {
    const int total = Nc * Kpad;
    convT_kernel<<<(total + 255) / 256, 256, 0, stream>>>(src, dst, K, Nc, Kpad, total);
  };
  conv(Wh0, Wh0T, H_DIM, G_DIM, H_DIM);
  conv(Wx0, Wx0Tp, F_DIM, G_DIM, 32);
  conv(Wx1, Wx1T, H_DIM, G_DIM, H_DIM);
  conv(Wh1, Wh1T, H_DIM, G_DIM, H_DIM);
  conv(Wt,  WtT,  H_DIM, H_DIM, H_DIM);
  conv(Wl,  WlT,  H_DIM, O_DIM, H_DIM);

  {
    const int total = N_BATCH * W_SEQ;
    build_inputs_kernel<<<(total + 255) / 256, 256, 0, stream>>>(x, noise, levels, seas, inputs);
  }
  {
    const int total = N_BATCH * O_DIM;
    build_labels_kernel<<<(total + 255) / 256, 256, 0, stream>>>(x, levels, seas, labels);
  }

  lstm1_kernel<<<(N_BATCH + 15) / 16, 384, 0, stream>>>(inputs, Wh0T, Wx0Tp, b0, h1_sel);
  lstm2_kernel<<<(N_BATCH + 15) / 16, 384, 0, stream>>>(h1_sel, Wx1T, Wh1T, b1, h2_last);
  proj_tanh_kernel<<<(N_BATCH + 15) / 16, 256, 0, stream>>>(h2_last, WtT, bt, tmp);
  proj_out_kernel<<<(N_BATCH + 15) / 16, 32, 0, stream>>>(tmp, WlT, bl, out);
}
